// MultiHeadAttention_79113297592453
// MI455X (gfx1250) — matrix operand order repaired and checked by reading, not compiled, not run on hardware
//
#include <hip/hip_runtime.h>
#include <hip/hip_bf16.h>

typedef __attribute__((ext_vector_type(16))) _Float16 v16h;
typedef __attribute__((ext_vector_type(2)))  _Float16 v2h;
typedef __attribute__((ext_vector_type(8)))  float    v8f;

#define S_LEN 2048
#define DM    1024
#define NH    16
#define DH    64
#define BATCH 2
#define MROWS (BATCH * S_LEN)   // 4096

// ---------------------------------------------------------------------------
// Fragment loaders, following CDNA5 ISA 7.12.2 VGPR layouts (wave32).
// A (16x32, f16): lane L holds row m = L&15.
//   VGPR j (j<4): K = 2*(j&3) + (L>=16 ? 8 : 0);  VGPR j>=4: +16.
// B (32x16, f16): lane L holds column n = L&15.
//   VGPR j: same K as A (2*(j&3) + (L>=16 ? 8 : 0), +16 for j>=4).   B is stored rows-indexed-by-n,
//   contiguous in K (i.e. W[n][k], k[n][d], vT[d][s]) so loads are v2h and
//   coalesce into b128.
// C/D (16x16, f32): lane L holds column n = L&15; VGPR r holds row
//   m = r + (L>=16 ? 8 : 0).
// ---------------------------------------------------------------------------
__device__ __forceinline__ v16h load_a_frag(const _Float16* base,
                                            int ld, int row0, int k0, int lane) {
  int m  = row0 + (lane & 15);
  int kh = (lane & 16) ? 8 : 0;
  const _Float16* p = base + (size_t)m * ld + k0;
  v16h a;
#pragma unroll
  for (int j = 0; j < 8; ++j) {
    int kb = ((j & 4) ? 16 : 0) + kh + 2 * (j & 3);
    v2h pr = *(const v2h*)(p + kb);
    a[2 * j]     = pr[0];
    a[2 * j + 1] = pr[1];
  }
  return a;
}

__device__ __forceinline__ v16h load_b_frag(const _Float16* base,
                                            int ld, int col0, int k0, int lane) {
  int n  = col0 + (lane & 15);
  int ko = (lane & 16) ? 8 : 0;
  const _Float16* p = base + (size_t)n * ld + k0 + ko;
  v16h b;
#pragma unroll
  for (int j = 0; j < 8; ++j) {
    v2h pr = *(const v2h*)(p + ((j & 4) ? 16 : 0) + 2 * (j & 3));
    b[2 * j]     = pr[0];
    b[2 * j + 1] = pr[1];
  }
  return b;
}

__device__ __forceinline__ v8f wmma_f16(v16h a, v16h b, v8f c) {
  return __builtin_amdgcn_wmma_f32_16x16x32_f16(false, a, false, b, (short)0, c,
                                                false, false);
}

// ---------------------------------------------------------------------------
// CDNA5 async global -> LDS copy (ASYNCcnt-tracked, no VGPR staging).
// lds_off is the byte offset within the workgroup LDS allocation (low 32 bits
// of a generic pointer to __shared__), gaddr a 64-bit global VA.
// ---------------------------------------------------------------------------
__device__ __forceinline__ void async_copy_b128(unsigned int lds_off,
                                                unsigned long long gaddr) {
  asm volatile("global_load_async_to_lds_b128 %0, %1, off"
               :: "v"(lds_off), "v"(gaddr) : "memory");
}
__device__ __forceinline__ void wait_async_le16() {
  asm volatile("s_wait_asynccnt 16" ::: "memory");
}
__device__ __forceinline__ void wait_async_0() {
  asm volatile("s_wait_asynccnt 0" ::: "memory");
}

// ---------------------------------------------------------------------------
// fp32 -> fp16 conversion
// ---------------------------------------------------------------------------
__global__ void cvt_f32_f16(const float* __restrict__ in,
                            _Float16* __restrict__ out, int n) {
  int i = blockIdx.x * blockDim.x + threadIdx.x;
  if (i < n) out[i] = (_Float16)in[i];
}

// ---------------------------------------------------------------------------
// C = A(f16, M x K row-major) @ W(f16, N x K row-major)^T + bias
// One wave computes a 64x32 tile (4x2 WMMA tiles): 8 WMMAs per K-step of 32
// with 6 fragment loads (B fragments reused 4x).
// mode 0: f16 out, per-head layout  [B,H,S,Dh]   (q, k)
// mode 1: f16 out, per-head transp. [B,H,Dh,S]   (v)
// mode 2: f32 out, row-major [M,N]               (final projection)
// ---------------------------------------------------------------------------
__global__ void __launch_bounds__(128)
gemm_xwT(const _Float16* __restrict__ A, const _Float16* __restrict__ W,
         const float* __restrict__ bias, void* __restrict__ out, int mode) {
  int lane = threadIdx.x & 31;
  int wid  = blockIdx.x * (blockDim.x >> 5) + (threadIdx.x >> 5);
  const int tilesN = DM / 32;                 // 32
  int m0 = (wid / tilesN) * 64;
  int n0 = (wid % tilesN) * 32;

  v8f c[4][2];
#pragma unroll
  for (int i = 0; i < 4; ++i)
#pragma unroll
    for (int j = 0; j < 2; ++j) c[i][j] = {};

  for (int kk = 0; kk < DM; kk += 32) {
    v16h b0 = load_b_frag(W, DM, n0,      kk, lane);
    v16h b1 = load_b_frag(W, DM, n0 + 16, kk, lane);
#pragma unroll
    for (int i = 0; i < 4; ++i) {
      v16h ai = load_a_frag(A, DM, m0 + i * 16, kk, lane);
      c[i][0] = wmma_f16(ai, b0, c[i][0]);
      c[i][1] = wmma_f16(ai, b1, c[i][1]);
    }
  }

  int rowoff = (lane & 16) ? 8 : 0;
  int nl     = lane & 15;
#pragma unroll
  for (int i = 0; i < 4; ++i) {
#pragma unroll
    for (int j = 0; j < 2; ++j) {
#pragma unroll
      for (int r = 0; r < 8; ++r) {
        int m = m0 + i * 16 + r + rowoff;
        int n = n0 + j * 16 + nl;
        float val = c[i][j][r] + bias[n];
        if (mode == 2) {
          ((float*)out)[(size_t)m * DM + n] = val;
        } else {
          int b  = m / S_LEN, s = m % S_LEN;
          int h  = n >> 6,    dh = n & 63;
          _Float16 hv = (_Float16)val;
          if (mode == 0)
            ((_Float16*)out)[(((size_t)(b * NH + h) * S_LEN + s) * DH) + dh] = hv;
          else
            ((_Float16*)out)[(((size_t)(b * NH + h) * DH + dh) * S_LEN) + s] = hv;
        }
      }
    }
  }
}

// ---------------------------------------------------------------------------
// Streaming-softmax (flash) attention. One wave = one (b, h, 16-row q tile).
// Key chunks of 32 columns are double-buffered in LDS via
// global_load_async_to_lds_b128 (16 x b128 per chunk: 4KB K rows + 4KB vT
// rows). While WMMAs consume chunk i from LDS, chunk i+1 streams in; the
// in-order ASYNCcnt lets `s_wait_asynccnt 16` gate on the current chunk only.
// ---------------------------------------------------------------------------
__global__ void __launch_bounds__(32)
flash_attn(const _Float16* __restrict__ q, const _Float16* __restrict__ k,
           const _Float16* __restrict__ vT, _Float16* __restrict__ concat) {
  __shared__ _Float16 kbuf[2][32 * 64];   // [local key s][d], ld = 64
  __shared__ _Float16 vbuf[2][64 * 32];   // [d][local key s], ld = 32
  __shared__ _Float16 pl[16 * 32];

  int lane = threadIdx.x & 31;
  int qt = blockIdx.x % (S_LEN / 16);
  int bh = blockIdx.x / (S_LEN / 16);
  int h  = bh % NH;
  int b  = bh / NH;

  const _Float16* qbase = q  + (size_t)bh * S_LEN * DH;
  const _Float16* kbase = k  + (size_t)bh * S_LEN * DH;
  const _Float16* vbase = vT + (size_t)bh * DH * S_LEN;
  int m0 = qt * 16;

  // Stage one 32-key chunk into LDS buffers: 8 b128 for K (contiguous 4KB)
  // + 8 b128 for vT (64 rows of 64B, 4 x 16B each). For both, the LDS byte
  // offset of 16B-chunk c is simply c*16.
  auto stage_chunk = [&](int buf, int j0) {
    unsigned int kdst = (unsigned int)(uintptr_t)&kbuf[buf][0];
    unsigned int vdst = (unsigned int)(uintptr_t)&vbuf[buf][0];
    const _Float16* ksrc = kbase + (size_t)j0 * DH;
#pragma unroll
    for (int i = 0; i < 8; ++i) {
      int c = i * 32 + lane;                       // 16-byte chunk id, 0..255
      async_copy_b128(kdst + c * 16,
                      (unsigned long long)(uintptr_t)(ksrc + c * 8));
    }
#pragma unroll
    for (int i = 0; i < 8; ++i) {
      int c = i * 32 + lane;
      int d = c >> 2, part = c & 3;
      async_copy_b128(vdst + c * 16,
                      (unsigned long long)(uintptr_t)(vbase + (size_t)d * S_LEN
                                                      + j0 + part * 8));
    }
  };

  // q fragments over Dh=64 (two K=32 chunks), pre-scaled by 1/sqrt(Dh)=0.125
  v16h qa0 = load_a_frag(qbase, DH, m0, 0,  lane);
  v16h qa1 = load_a_frag(qbase, DH, m0, 32, lane);
#pragma unroll
  for (int e = 0; e < 16; ++e) {
    qa0[e] = qa0[e] * (_Float16)0.125f;
    qa1[e] = qa1[e] * (_Float16)0.125f;
  }

  v8f acc[4];
#pragma unroll
  for (int t = 0; t < 4; ++t) acc[t] = {};
  float rmax[8], rsum[8];
#pragma unroll
  for (int r = 0; r < 8; ++r) { rmax[r] = -1e30f; rsum[r] = 0.0f; }

  int rowoff = (lane & 16) ? 8 : 0;
  int nl     = lane & 15;

  stage_chunk(0, 0);

  for (int j0 = 0; j0 < S_LEN; j0 += 32) {
    int cur = (j0 >> 5) & 1;
    if (j0 + 32 < S_LEN) {
      stage_chunk(cur ^ 1, j0 + 32);   // overlap next chunk with compute
      wait_async_le16();               // in-order: current chunk has landed
    } else {
      wait_async_0();
    }

    // score tiles from LDS: columns [0,16) and [16,32) of this chunk
    const _Float16* kl = &kbuf[cur][0];
    v16h kb00 = load_b_frag(kl, DH, 0,  0,  lane);
    v16h kb01 = load_b_frag(kl, DH, 0,  32, lane);
    v16h kb10 = load_b_frag(kl, DH, 16, 0,  lane);
    v16h kb11 = load_b_frag(kl, DH, 16, 32, lane);
    v8f s0 = {}, s1 = {};
    s0 = wmma_f16(qa0, kb00, s0);
    s0 = wmma_f16(qa1, kb01, s0);
    s1 = wmma_f16(qa0, kb10, s1);
    s1 = wmma_f16(qa1, kb11, s1);

    // online softmax update (per-row stats live in each 16-lane half)
    float alpha[8];
#pragma unroll
    for (int r = 0; r < 8; ++r) {
      float mt = fmaxf(s0[r], s1[r]);
#pragma unroll
      for (int msk = 1; msk < 16; msk <<= 1)
        mt = fmaxf(mt, __shfl_xor(mt, msk, 32));
      float mnew = fmaxf(rmax[r], mt);
      float al   = __expf(rmax[r] - mnew);
      rmax[r]    = mnew;
      float p0 = __expf(s0[r] - mnew);
      float p1 = __expf(s1[r] - mnew);
      float ps = p0 + p1;
#pragma unroll
      for (int msk = 1; msk < 16; msk <<= 1)
        ps += __shfl_xor(ps, msk, 32);
      rsum[r] = rsum[r] * al + ps;
      s0[r] = p0;
      s1[r] = p1;
      alpha[r] = al;
    }
#pragma unroll
    for (int t = 0; t < 4; ++t)
#pragma unroll
      for (int r = 0; r < 8; ++r) acc[t][r] *= alpha[r];

    // P tile (C layout) -> LDS (16x32 f16 row-major) -> A fragment
    __syncthreads();
#pragma unroll
    for (int r = 0; r < 8; ++r) {
      pl[(r + rowoff) * 32 + nl]      = (_Float16)s0[r];
      pl[(r + rowoff) * 32 + 16 + nl] = (_Float16)s1[r];
    }
    __syncthreads();
    v16h pa = load_a_frag(pl, 32, 0, 0, lane);

    // acc += P @ V   (vbuf rows are d, contiguous in local s)
    const _Float16* vl = &vbuf[cur][0];
#pragma unroll
    for (int t = 0; t < 4; ++t) {
      v16h vb = load_b_frag(vl + t * 16 * 32, 32, 0, 0, lane);
      acc[t] = wmma_f16(pa, vb, acc[t]);
    }
  }

  // normalize and write concat [B,S,D] f16
  _Float16* obase = concat + ((size_t)b * S_LEN + m0) * DM + h * DH;
#pragma unroll
  for (int r = 0; r < 8; ++r) {
    float inv = 1.0f / rsum[r];
    int m = r + rowoff;
#pragma unroll
    for (int t = 0; t < 4; ++t)
      obase[(size_t)m * DM + t * 16 + nl] = (_Float16)(acc[t][r] * inv);
  }
}

// ---------------------------------------------------------------------------
extern "C" void kernel_launch(void* const* d_in, const int* in_sizes, int n_in,
                              void* d_out, int out_size, void* d_ws, size_t ws_size,
                              hipStream_t stream) {
  const float* Q  = (const float*)d_in[0];
  const float* K  = (const float*)d_in[1];
  const float* V  = (const float*)d_in[2];
  const float* Wq = (const float*)d_in[3];
  const float* Wk = (const float*)d_in[4];
  const float* Wv = (const float*)d_in[5];
  const float* Wo = (const float*)d_in[6];
  const float* bq = (const float*)d_in[7];
  const float* bk = (const float*)d_in[8];
  const float* bv = (const float*)d_in[9];
  const float* bo = (const float*)d_in[10];

  const size_t SZ_X = (size_t)MROWS * DM;   // 4096*1024 elements
  const size_t SZ_W = (size_t)DM * DM;      // 1024*1024 elements

  char* ws = (char*)d_ws;
  size_t cur = 0;
  auto carve = [&](size_t elems) {
    _Float16* p = (_Float16*)(ws + cur);
    cur += elems * sizeof(_Float16);
    return p;
  };
  _Float16* Q16   = carve(SZ_X);
  _Float16* K16   = carve(SZ_X);
  _Float16* V16   = carve(SZ_X);
  _Float16* Wq16  = carve(SZ_W);
  _Float16* Wk16  = carve(SZ_W);
  _Float16* Wv16  = carve(SZ_W);
  _Float16* Wo16  = carve(SZ_W);
  _Float16* q16   = carve(SZ_X);   // [B,H,S,Dh]
  _Float16* k16   = carve(SZ_X);   // [B,H,S,Dh]
  _Float16* vT16  = carve(SZ_X);   // [B,H,Dh,S]
  _Float16* cat16 = carve(SZ_X);   // [B,S,D]
  (void)ws_size;                   // ~64 MB required

  // fp32 -> fp16 conversions
  {
    int nx = (int)SZ_X, nw = (int)SZ_W;
    dim3 blk(256);
    cvt_f32_f16<<<(nx + 255) / 256, blk, 0, stream>>>(Q,  Q16,  nx);
    cvt_f32_f16<<<(nx + 255) / 256, blk, 0, stream>>>(K,  K16,  nx);
    cvt_f32_f16<<<(nx + 255) / 256, blk, 0, stream>>>(V,  V16,  nx);
    cvt_f32_f16<<<(nw + 255) / 256, blk, 0, stream>>>(Wq, Wq16, nw);
    cvt_f32_f16<<<(nw + 255) / 256, blk, 0, stream>>>(Wk, Wk16, nw);
    cvt_f32_f16<<<(nw + 255) / 256, blk, 0, stream>>>(Wv, Wv16, nw);
    cvt_f32_f16<<<(nw + 255) / 256, blk, 0, stream>>>(Wo, Wo16, nw);
  }

  // projection GEMMs: (MROWS/64)*(DM/32) = 2048 waves, 4 waves/block
  const int gemm_blocks = (MROWS / 64) * (DM / 32) / 4;  // 512
  gemm_xwT<<<gemm_blocks, 128, 0, stream>>>(Q16, Wq16, bq, (void*)q16,  0);
  gemm_xwT<<<gemm_blocks, 128, 0, stream>>>(K16, Wk16, bk, (void*)k16,  0);
  gemm_xwT<<<gemm_blocks, 128, 0, stream>>>(V16, Wv16, bv, (void*)vT16, 1);

  // flash attention: one wave per (b, h, 16-row query tile)
  flash_attn<<<BATCH * NH * (S_LEN / 16), 32, 0, stream>>>(q16, k16, vT16, cat16);

  // output projection -> fp32 d_out
  gemm_xwT<<<gemm_blocks, 128, 0, stream>>>(cat16, Wo16, bo, d_out, 2);
  (void)in_sizes; (void)n_in; (void)out_size;
}
